// _PagedMoEFFN_30992484008193
// MI455X (gfx1250) — matrix operand order repaired and checked by reading, not compiled, not run on hardware
//
#include <hip/hip_runtime.h>
#include <hip/hip_bf16.h>
#include <stdint.h>

#define T_  2048
#define D_  768
#define I_  2048
#define E_  8
#define MT_  (T_ / 16)  // fallback: 16-row tiles
#define MT2_ (T_ / 32)  // fast path: 32-row tiles
#define X2LD 768        // fast path x-tile row stride (halves), XOR-swizzled
#define H2LD 256        // fast path h-slice width (halves); 8 slices cover I_

typedef __attribute__((ext_vector_type(16))) _Float16 v16h;
typedef __attribute__((ext_vector_type(8)))  _Float16 h8;
typedef __attribute__((ext_vector_type(8)))  float    v8f;
typedef __attribute__((ext_vector_type(4)))  float    f4;
typedef __attribute__((ext_vector_type(4)))  int      i4;

// ---------------------------------------------------------------------------
// WMMA wrapper: D = A(16x32 f16) * B(32x16 f16) + C(16x16 f32)
// ---------------------------------------------------------------------------
__device__ __forceinline__ v8f wmma16(v16h a, v16h b, v8f c) {
  return __builtin_amdgcn_wmma_f32_16x16x32_f16(
      false, a, false, b, (short)0, c, false, false);
}

// ---------------------------------------------------------------------------
// Fast-path loaders (all-f16 operands)
// B layout: lanes 0-15 hold K=k0..+7,k0+16..+23; lanes 16-31 K=+8..+15,+24..+31.
// ---------------------------------------------------------------------------
__device__ __forceinline__ v16h load_b_h(const _Float16* __restrict__ base,
                                         int ld, int col, int k0) {
  const int lane = threadIdx.x & 31;
  const _Float16* p = base + (size_t)col * ld + k0 + ((lane >> 4) << 3);
  h8 b0 = *(const h8*)p;
  h8 b1 = *(const h8*)(p + 16);
  v16h f;
#pragma unroll
  for (int i = 0; i < 8; ++i) { f[i] = b0[i]; f[8 + i] = b1[i]; }
  return f;
}

// A from the swizzled LDS x tile (row in [0,32), stride X2LD, xor-skewed).
// A layout: lanes 0-15 hold K=k0..+7,k0+16..+23; lanes 16-31 K=+8..+15,+24..+31.
__device__ __forceinline__ v16h load_a_x(const _Float16* xb, int row, int kb) {
  const int lane = threadIdx.x & 31;
  const int klo = kb + ((lane >> 4) << 3);
  const int sw = (row & 15) << 3;
  h8 a0 = *(const h8*)&xb[row * X2LD + (klo ^ sw)];
  h8 a1 = *(const h8*)&xb[row * X2LD + ((klo + 16) ^ sw)];
  v16h f;
#pragma unroll
  for (int i = 0; i < 8; ++i) { f[i] = a0[i]; f[8 + i] = a1[i]; }
  return f;
}

// ---------------------------------------------------------------------------
// Fallback loaders (convert in-register; used only when ws is small)
// ---------------------------------------------------------------------------
__device__ __forceinline__ v16h load_a_f32(const float* __restrict__ base,
                                           int ld, int row, int k0) {
  const int lane = threadIdx.x & 31;
  const float* p = base + (size_t)row * ld + k0 + ((lane >> 4) << 3);
  f4 a0 = *(const f4*)(p);
  f4 a1 = *(const f4*)(p + 4);
  f4 b0 = *(const f4*)(p + 16);
  f4 b1 = *(const f4*)(p + 20);
  v16h f;
#pragma unroll
  for (int i = 0; i < 4; ++i) {
    f[i] = (_Float16)a0[i];  f[4 + i] = (_Float16)a1[i];
    f[8 + i] = (_Float16)b0[i]; f[12 + i] = (_Float16)b1[i];
  }
  return f;
}

__device__ __forceinline__ v16h load_b_w(const float* __restrict__ base,
                                         int ld, int col, int k0) {
  const int lane = threadIdx.x & 31;
  const float* p = base + (size_t)col * ld + k0 + ((lane >> 4) << 3);
  f4 a0 = *(const f4*)(p);
  f4 a1 = *(const f4*)(p + 4);
  f4 a2 = *(const f4*)(p + 16);
  f4 a3 = *(const f4*)(p + 20);
  v16h f;
#pragma unroll
  for (int i = 0; i < 4; ++i) {
    f[i] = (_Float16)a0[i];  f[4 + i] = (_Float16)a1[i];
    f[8 + i] = (_Float16)a2[i]; f[12 + i] = (_Float16)a3[i];
  }
  return f;
}

__device__ __forceinline__ v16h load_b_w(const int* __restrict__ base,
                                         int ld, int col, int k0) {
  const int lane = threadIdx.x & 31;
  const int* p = base + (size_t)col * ld + k0 + ((lane >> 4) << 3);
  i4 a0 = *(const i4*)(p);
  i4 a1 = *(const i4*)(p + 4);
  i4 a2 = *(const i4*)(p + 16);
  i4 a3 = *(const i4*)(p + 20);
  v16h f;
#pragma unroll
  for (int i = 0; i < 4; ++i) {
    f[i] = (_Float16)(float)a0[i];  f[4 + i] = (_Float16)(float)a1[i];
    f[8 + i] = (_Float16)(float)a2[i]; f[12 + i] = (_Float16)(float)a3[i];
  }
  return f;
}

// ---------------------------------------------------------------------------
__global__ void k_init_cnt(int* __restrict__ cnt) {
  if (threadIdx.x < 16) cnt[threadIdx.x] = 0;
}

// ---------------------------------------------------------------------------
// Router: one wave per token; softmax -> top-2 -> renormalize -> append.
// ---------------------------------------------------------------------------
__global__ void __launch_bounds__(256)
k_router(const float* __restrict__ x, const float* __restrict__ rw,
         int* __restrict__ cnt, int* __restrict__ tokidx,
         float* __restrict__ tokw) {
  const int wave = threadIdx.x >> 5;
  const int lane = threadIdx.x & 31;
  const int t = blockIdx.x * 8 + wave;

  float acc[E_];
#pragma unroll
  for (int e = 0; e < E_; ++e) acc[e] = 0.f;
  for (int d = lane; d < D_; d += 32) {
    float xv = x[(size_t)t * D_ + d];
#pragma unroll
    for (int e = 0; e < E_; ++e) acc[e] += xv * rw[(size_t)e * D_ + d];
  }
#pragma unroll
  for (int e = 0; e < E_; ++e) {
#pragma unroll
    for (int m = 16; m >= 1; m >>= 1) acc[e] += __shfl_xor(acc[e], m, 32);
  }
  if (lane == 0) {
    float mx = acc[0];
#pragma unroll
    for (int e = 1; e < E_; ++e) mx = fmaxf(mx, acc[e]);
    float p[E_];
#pragma unroll
    for (int e = 0; e < E_; ++e) p[e] = __expf(acc[e] - mx);
    int e0 = 0;
#pragma unroll
    for (int e = 1; e < E_; ++e) if (p[e] > p[e0]) e0 = e;
    int e1 = (e0 == 0) ? 1 : 0;
#pragma unroll
    for (int e = 0; e < E_; ++e) if (e != e0 && p[e] > p[e1]) e1 = e;
    float inv = 1.f / (p[e0] + p[e1]);
    int p0 = atomicAdd(&cnt[e0], 1);
    tokidx[e0 * T_ + p0] = t; tokw[e0 * T_ + p0] = p[e0] * inv;
    int p1 = atomicAdd(&cnt[e1], 1);
    tokidx[e1 * T_ + p1] = t; tokw[e1 * T_ + p1] = p[e1] * inv;
  }
}

// ---------------------------------------------------------------------------
// Dequant passes (fast path): convert once, reuse many times per expert.
// ---------------------------------------------------------------------------
__global__ void __launch_bounds__(256)
k_dq_f32(const float* __restrict__ in, _Float16* __restrict__ outp) {
  const size_t i = ((size_t)blockIdx.x * blockDim.x + threadIdx.x) * 8;
  f4 a = *(const f4*)(in + i);
  f4 b = *(const f4*)(in + i + 4);
  h8 o;
#pragma unroll
  for (int k = 0; k < 4; ++k) { o[k] = (_Float16)a[k]; o[4 + k] = (_Float16)b[k]; }
  *(h8*)(outp + i) = o;
}

__global__ void __launch_bounds__(256)
k_dq_tern(const int* __restrict__ q, const float* __restrict__ scales, int scol,
          size_t per_e, _Float16* __restrict__ outp) {
  const size_t i = ((size_t)blockIdx.x * blockDim.x + threadIdx.x) * 8;
  const int e = (int)(i / per_e);
  const float s = scales[e * 3 + scol];   // bake dequant scale into f16 weight
  i4 a = *(const i4*)(q + i);
  i4 b = *(const i4*)(q + i + 4);
  h8 o;
#pragma unroll
  for (int k = 0; k < 4; ++k) {
    o[k]     = (_Float16)(s * (float)a[k]);
    o[4 + k] = (_Float16)(s * (float)b[k]);
  }
  *(h8*)(outp + i) = o;
}

// ---------------------------------------------------------------------------
// FAST PATH: fused SwiGLU expert block, all-f16 operands, 32-token tiles.
//  - async-DMA 32 gathered x rows into swizzled LDS (48 KB)
//  - h computed in eight 256-wide slices of I (16 KB LDS); weights read once
//  - every B fragment feeds two A row-tiles => ~1 global b128 per WMMA
//  - scales baked into weights; routing weight applied at scatter
// ---------------------------------------------------------------------------
template <bool SHARED>
__global__ void __launch_bounds__(256)
k_moe_f16(const _Float16* __restrict__ xh,
          const _Float16* __restrict__ Wg, const _Float16* __restrict__ Wu,
          const _Float16* __restrict__ Wd,
          const int* __restrict__ cnt, const int* __restrict__ tokidx,
          const float* __restrict__ tokw,
          float* __restrict__ out) {
  __shared__ _Float16 xbuf[32 * X2LD];   // 49152 B, XOR-swizzled
  __shared__ _Float16 hbuf[32 * H2LD];   // 16384 B, XOR-swizzled  (total 64 KB)

  const int e  = SHARED ? 0 : (blockIdx.x / MT2_);
  const int mt = SHARED ? blockIdx.x : (blockIdx.x % MT2_);

  int n = T_;
  if (!SHARED) n = cnt[e];
  if (mt * 32 >= n) return;            // block-uniform exit (EXEC stays full)

  const _Float16* wg = Wg;
  const _Float16* wu = Wu;
  const _Float16* wd = Wd;
  if (!SHARED) {
    wg = Wg + (size_t)e * I_ * D_;
    wu = Wu + (size_t)e * I_ * D_;
    wd = Wd + (size_t)e * D_ * I_;
  }

  const int tid  = threadIdx.x;
  const int wave = tid >> 5;
  const int lane = tid & 31;
  const int li   = lane & 15;

  // token index / routing weight for a row in [0,32)
  auto rowtok = [&](int row) -> int {
    const int idxm = mt * 32 + row;
    if (SHARED) return idxm;
    const int cl = (idxm < n) ? idxm : (n - 1);
    return tokidx[e * T_ + cl];
  };
  auto roww = [&](int row) -> float {
    if (SHARED) return 1.f;
    const int idxm = mt * 32 + row;
    return (idxm < n) ? tokw[e * T_ + idxm] : 0.f;
  };

  const int   myrow0 = rowtok(li);        // lane li owns rows li and 16+li
  const int   myrow1 = rowtok(16 + li);
  const float myw0   = roww(li);
  const float myw1   = roww(16 + li);

  // --- stage 32 gathered x rows (f16) into LDS via async DMA ---
  // 32 rows x 768 halves = 3072 x 16B chunks; 256 threads x 12 iters.
  {
#pragma unroll
    for (int it = 0; it < 12; ++it) {
      const int c   = tid + 256 * it;      // chunk id 0..3071
      const int row = c / 96;
      const int o16 = c % 96;
      const int tok = rowtok(row);
      const unsigned long long ga =
          (unsigned long long)(const void*)xh +
          ((unsigned long long)tok * D_ + (unsigned long long)o16 * 8) * 2ull;
      const unsigned int lo = (unsigned int)(uintptr_t)
          &xbuf[row * X2LD + ((o16 * 8) ^ ((row & 15) << 3))];
      asm volatile("global_load_async_to_lds_b128 %0, %1, off"
                   :: "v"(lo), "v"(ga) : "memory");
    }
    asm volatile("s_wait_asynccnt 0x0" ::: "memory");
  }
  __syncthreads();

  const int cb = wave * 96;                // phase-2 cols: 8 waves cover D_
  v8f ac[2][6];
#pragma unroll
  for (int rt = 0; rt < 2; ++rt)
#pragma unroll
    for (int j = 0; j < 6; ++j)
#pragma unroll
      for (int v = 0; v < 8; ++v) ac[rt][j][v] = 0.f;

  for (int e8 = 0; e8 < 8; ++e8) {         // eight 256-wide slices of I
    // ----- Phase 1: gate/up GEMMs + SwiGLU -> LDS (this slice) -----
    const int c0 = wave * 32;              // local col base in [0,256)
    v8f ag[2][2], au[2][2];
#pragma unroll
    for (int rt = 0; rt < 2; ++rt)
#pragma unroll
      for (int j = 0; j < 2; ++j)
#pragma unroll
        for (int v = 0; v < 8; ++v) { ag[rt][j][v] = 0.f; au[rt][j][v] = 0.f; }

    for (int kb = 0; kb < D_; kb += 32) {
      v16h af0 = load_a_x(xbuf, li, kb);
      v16h af1 = load_a_x(xbuf, 16 + li, kb);
#pragma unroll
      for (int j = 0; j < 2; ++j) {
        const int icol = e8 * H2LD + c0 + j * 16 + li;
        v16h bg = load_b_h(wg, D_, icol, kb);
        ag[0][j] = wmma16(af0, bg, ag[0][j]);
        ag[1][j] = wmma16(af1, bg, ag[1][j]);
        v16h bu = load_b_h(wu, D_, icol, kb);
        au[0][j] = wmma16(af0, bu, au[0][j]);
        au[1][j] = wmma16(af1, bu, au[1][j]);
      }
    }
#pragma unroll
    for (int rt = 0; rt < 2; ++rt) {
#pragma unroll
      for (int j = 0; j < 2; ++j) {
#pragma unroll
        for (int v = 0; v < 8; ++v) {
          const int m  = v + ((lane >> 4) << 3);
          const int m2 = m + 16 * rt;
          const int c2 = c0 + j * 16 + li;
          const float gv = ag[rt][j][v];
          const float uv = au[rt][j][v];
          const float hv = (gv / (1.f + __expf(-gv))) * uv;   // silu(g)*u
          hbuf[m2 * H2LD + (c2 ^ ((m2 & 15) << 3))] = (_Float16)hv;
        }
      }
    }
    __syncthreads();

    // ----- Phase 2: down GEMM partial accumulation (this slice) -----
    const int sw = li << 3;
    for (int kb = 0; kb < H2LD; kb += 32) {
      const int klo = kb + ((lane >> 4) << 3);
      h8 a00 = *(const h8*)&hbuf[li * H2LD + (klo ^ sw)];
      h8 a01 = *(const h8*)&hbuf[li * H2LD + ((klo + 16) ^ sw)];
      h8 a10 = *(const h8*)&hbuf[(16 + li) * H2LD + (klo ^ sw)];
      h8 a11 = *(const h8*)&hbuf[(16 + li) * H2LD + ((klo + 16) ^ sw)];
      v16h af0, af1;
#pragma unroll
      for (int i = 0; i < 8; ++i) {
        af0[i] = a00[i]; af0[8 + i] = a01[i];
        af1[i] = a10[i]; af1[8 + i] = a11[i];
      }
#pragma unroll
      for (int j = 0; j < 6; ++j) {
        v16h bf = load_b_h(wd, I_, cb + j * 16 + li, e8 * H2LD + kb);
        ac[0][j] = wmma16(af0, bf, ac[0][j]);
        ac[1][j] = wmma16(af1, bf, ac[1][j]);
      }
    }
    __syncthreads();                       // hbuf reused by next slice
  }

  // ----- scatter -----
#pragma unroll
  for (int rt = 0; rt < 2; ++rt) {
#pragma unroll
    for (int j = 0; j < 6; ++j) {
#pragma unroll
      for (int v = 0; v < 8; ++v) {
        const int m   = v + ((lane >> 4) << 3);
        const int col = cb + j * 16 + li;
        const int   tr = (rt == 0) ? __shfl(myrow0, m, 32) : __shfl(myrow1, m, 32);
        const float tw = (rt == 0) ? __shfl(myw0, m, 32)   : __shfl(myw1, m, 32);
        const float val = tw * ac[rt][j][v];
        if (SHARED) out[(size_t)tr * D_ + col] = val;
        else        atomicAdd(&out[(size_t)tr * D_ + col], val);
      }
    }
  }
}

// ---------------------------------------------------------------------------
// FALLBACK PATH (small ws): converts operands in-register, 16-row tiles.
// ---------------------------------------------------------------------------
template <bool SHARED, typename TW>
__global__ void __launch_bounds__(256)
k_moe_fb(const float* __restrict__ x,
         const TW* __restrict__ Wg, const TW* __restrict__ Wu,
         const TW* __restrict__ Wd,
         const float* __restrict__ scales,
         const int* __restrict__ cnt, const int* __restrict__ tokidx,
         const float* __restrict__ tokw,
         float* __restrict__ out) {
  __shared__ _Float16 hbuf[16 * I_];   // 65536 bytes

  const int e  = SHARED ? 0 : (blockIdx.x / MT_);
  const int mt = SHARED ? blockIdx.x : (blockIdx.x % MT_);
  int n = T_;
  if (!SHARED) n = cnt[e];
  if (mt * 16 >= n) return;

  float s0 = 1.f, s1 = 1.f, s2 = 1.f;
  const TW* wg = Wg; const TW* wu = Wu; const TW* wd = Wd;
  if (!SHARED) {
    s0 = scales[e * 3 + 0]; s1 = scales[e * 3 + 1]; s2 = scales[e * 3 + 2];
    wg = Wg + (size_t)e * I_ * D_;
    wu = Wu + (size_t)e * I_ * D_;
    wd = Wd + (size_t)e * D_ * I_;
  }

  const int tid = threadIdx.x;
  const int wave = tid >> 5;
  const int lane = tid & 31;
  const int li = lane & 15;

  int myrow; float myw;
  {
    const int idxm = mt * 16 + li;
    if (SHARED) { myrow = idxm; myw = 1.f; }
    else {
      const int cl = (idxm < n) ? idxm : (n - 1);
      myrow = tokidx[e * T_ + cl];
      myw = (idxm < n) ? tokw[e * T_ + idxm] : 0.f;
    }
  }

  const int nb0 = wave * 256;
  for (int g = 0; g < 4; ++g) {
    const int ncol = nb0 + g * 64;
    v8f ag[4], au[4];
#pragma unroll
    for (int j = 0; j < 4; ++j)
#pragma unroll
      for (int v = 0; v < 8; ++v) { ag[j][v] = 0.f; au[j][v] = 0.f; }
    for (int kb = 0; kb < D_; kb += 32) {
      v16h af = load_a_f32(x, D_, myrow, kb);
#pragma unroll
      for (int j = 0; j < 4; ++j) {
        v16h bg = load_b_w(wg, D_, ncol + j * 16 + li, kb);
        ag[j] = wmma16(af, bg, ag[j]);
        v16h bu = load_b_w(wu, D_, ncol + j * 16 + li, kb);
        au[j] = wmma16(af, bu, au[j]);
      }
    }
#pragma unroll
    for (int j = 0; j < 4; ++j)
#pragma unroll
      for (int v = 0; v < 8; ++v) {
        const int m = v + ((lane >> 4) << 3);
        const int col = ncol + j * 16 + li;
        const float gv = ag[j][v] * s0;
        const float uv = au[j][v] * s1;
        const float hv = (gv / (1.f + __expf(-gv))) * uv;
        hbuf[m * I_ + (col ^ (m << 3))] = (_Float16)hv;
      }
  }
  __syncthreads();

  const int cb = wave * 96;
  v8f ac[6];
#pragma unroll
  for (int j = 0; j < 6; ++j)
#pragma unroll
    for (int v = 0; v < 8; ++v) ac[j][v] = 0.f;

  const int roff = li * I_;
  const int sw = li << 3;
  for (int kb = 0; kb < I_; kb += 32) {
    const int klo = kb + ((lane >> 4) << 3);
    h8 a0 = *(const h8*)&hbuf[roff + (klo ^ sw)];
    h8 a1 = *(const h8*)&hbuf[roff + ((klo + 16) ^ sw)];
    v16h af;
#pragma unroll
    for (int i = 0; i < 8; ++i) { af[i] = a0[i]; af[8 + i] = a1[i]; }
#pragma unroll
    for (int j = 0; j < 6; ++j) {
      v16h bf = load_b_w(wd, I_, cb + j * 16 + li, kb);
      ac[j] = wmma16(af, bf, ac[j]);
    }
  }
#pragma unroll
  for (int j = 0; j < 6; ++j)
#pragma unroll
    for (int v = 0; v < 8; ++v) {
      const int m = v + ((lane >> 4) << 3);
      const int col = cb + j * 16 + li;
      const int tr = __shfl(myrow, m, 32);
      const float tw = __shfl(myw, m, 32);
      const float val = tw * s2 * ac[j][v];
      if (SHARED) out[(size_t)tr * D_ + col] = val;
      else        atomicAdd(&out[(size_t)tr * D_ + col], val);
    }
}

// ---------------------------------------------------------------------------
extern "C" void kernel_launch(void* const* d_in, const int* in_sizes, int n_in,
                              void* d_out, int out_size, void* d_ws,
                              size_t ws_size, hipStream_t stream) {
  const float* x  = (const float*)d_in[0];
  const float* rw = (const float*)d_in[1];
  const float* sg = (const float*)d_in[2];
  const float* su = (const float*)d_in[3];
  const float* sd = (const float*)d_in[4];
  const int*   qg = (const int*)d_in[5];
  const int*   qu = (const int*)d_in[6];
  const int*   qd = (const int*)d_in[7];
  const float* sc = (const float*)d_in[8];
  float* out = (float*)d_out;

  uint8_t* ws = (uint8_t*)d_ws;
  size_t off = 0;
  auto carve = [&](size_t bytes) -> void* {
    void* p = ws + off;
    off = (off + bytes + 255) & ~(size_t)255;
    return p;
  };

  int*   cnt    = (int*)carve(16 * sizeof(int));
  int*   tokidx = (int*)carve((size_t)E_ * T_ * sizeof(int));
  float* tokw   = (float*)carve((size_t)E_ * T_ * sizeof(float));

  _Float16* xh  = (_Float16*)carve((size_t)T_ * D_ * 2);
  _Float16* sgh = (_Float16*)carve((size_t)I_ * D_ * 2);
  _Float16* suh = (_Float16*)carve((size_t)I_ * D_ * 2);
  _Float16* sdh = (_Float16*)carve((size_t)D_ * I_ * 2);
  _Float16* wgh = (_Float16*)carve((size_t)E_ * I_ * D_ * 2);
  _Float16* wuh = (_Float16*)carve((size_t)E_ * I_ * D_ * 2);
  _Float16* wdh = (_Float16*)carve((size_t)E_ * D_ * I_ * 2);
  const size_t big_need = off;

  k_init_cnt<<<1, 32, 0, stream>>>(cnt);
  k_router<<<T_ / 8, 256, 0, stream>>>(x, rw, cnt, tokidx, tokw);

  if (ws_size >= big_need) {
    // one-shot dequant to f16 (scales baked); then all-f16 WMMA path
    k_dq_f32<<<(T_ * D_ / 8) / 256, 256, 0, stream>>>(x, xh);
    k_dq_f32<<<(I_ * D_ / 8) / 256, 256, 0, stream>>>(sg, sgh);
    k_dq_f32<<<(I_ * D_ / 8) / 256, 256, 0, stream>>>(su, suh);
    k_dq_f32<<<(D_ * I_ / 8) / 256, 256, 0, stream>>>(sd, sdh);
    const int tb = (E_ * I_ * D_ / 8) / 256;   // 6144
    k_dq_tern<<<tb, 256, 0, stream>>>(qg, sc, 0, (size_t)I_ * D_, wgh);
    k_dq_tern<<<tb, 256, 0, stream>>>(qu, sc, 1, (size_t)I_ * D_, wuh);
    k_dq_tern<<<tb, 256, 0, stream>>>(qd, sc, 2, (size_t)D_ * I_, wdh);

    k_moe_f16<true><<<MT2_, 256, 0, stream>>>(
        xh, sgh, suh, sdh, nullptr, nullptr, nullptr, out);
    k_moe_f16<false><<<E_ * MT2_, 256, 0, stream>>>(
        xh, wgh, wuh, wdh, cnt, tokidx, tokw, out);
  } else {
    // fallback: convert in-register
    k_moe_fb<true, float><<<MT_, 256, 0, stream>>>(
        x, sg, su, sd, nullptr, nullptr, nullptr, nullptr, out);
    k_moe_fb<false, int><<<E_ * MT_, 256, 0, stream>>>(
        x, qg, qu, qd, sc, cnt, tokidx, tokw, out);
  }
}
